// RSNN_89627377533334
// MI455X (gfx1250) — matrix operand order repaired and checked by reading, not compiled, not run on hardware
//
#include <hip/hip_runtime.h>
#include <hip/hip_bf16.h>
#include <stdint.h>

// ---------------- problem constants (from reference) ----------------
constexpr int T    = 100;
constexpr int B    = 128;
constexpr int NIN  = 512;
constexpr int NS   = 1024;
constexpr int NA   = 1024;
constexpr int NOUT = 256;
constexpr int NH   = NS + NA;       // 2048 recurrent neurons
constexpr int KIN  = NIN + NH;      // 2560 fan-in of hidden GEMM
constexpr int SIM  = 2;

#define ALPHA     0.9f
#define THRESH    1.0f
#define BETA      1.8f
#define RHO       0.995f
#define ALPHA_OUT 0.9f

// LDS row stride (u16 units) for staged weight tiles: 40 u16 = 80 bytes.
// 16-byte aligned (b128 ok) and 20*r mod 64 gives 16 distinct bank starts
// over the 16 rows a fragment read touches -> conflict-free ds_load_b128.
#define BSTRIDE 40

#define AS1 __attribute__((address_space(1)))
#define AS3 __attribute__((address_space(3)))

// ---------------- types ----------------
typedef unsigned short u16;
typedef int v4i __attribute__((vector_size(16)));     // matches async-LDS builtin param
typedef __attribute__((ext_vector_type(8)))  u16    u16x8;
typedef __attribute__((ext_vector_type(16))) __bf16 bf16x16;
typedef __attribute__((ext_vector_type(8)))  float  f32x8;

union Frag { u16x8 h[2]; bf16x16 bf; };

__device__ __forceinline__ u16 cvt_bf16_rne(float f) {
  union { float f; unsigned int u; } c; c.f = f;
  unsigned int u = c.u;
  u += 0x7FFFu + ((u >> 16) & 1u);          // round-to-nearest-even
  return (u16)(u >> 16);
}

__device__ __forceinline__ f32x8 wmma_bf16(const Frag& a, const Frag& b, f32x8 c) {
  // D = A(16x32 bf16) x B(32x16 bf16) + C(16x16 f32)
  return __builtin_amdgcn_wmma_f32_16x16x32_bf16(
      /*neg_a=*/false, a.bf, /*neg_b=*/false, b.bf,
      /*c_mod=*/(short)0, c, /*reuse_a=*/false, /*reuse_b=*/false);
}

__device__ __forceinline__ void wait_async0() {
#if __has_builtin(__builtin_amdgcn_s_wait_asynccnt)
  __builtin_amdgcn_s_wait_asynccnt(0);
#else
  asm volatile("s_wait_asynccnt 0x0" ::: "memory");
#endif
}

// ---------------- f32 -> bf16 conversion (once per call; weights stay L2-resident) --------
__global__ void cvt_f32_bf16(const float* __restrict__ src, u16* __restrict__ dst, int n) {
  int i = blockIdx.x * blockDim.x + threadIdx.x;
  int stride = gridDim.x * blockDim.x;
  for (; i < n; i += stride) dst[i] = cvt_bf16_rne(src[i]);
}

// ---------------- hidden GEMM + fused LIF/adaptive-LIF epilogue ----------------
// H = [x_t | spk_in] @ [Ws;Wa]^T   -> 128 x 2048, then per-neuron state update.
// grid (2, 32); block 256 = 8 waves (4 M-waves x 2 N-waves); wave tile 16M x 32N.
// Weight tiles (64 rows x 32 k, 4KB) are staged block-wide into LDS, double
// buffered, via CDNA5 async global->LDS loads (ASYNCcnt) when available.
__global__ __launch_bounds__(256) void rsnn_hidden(
    const u16* __restrict__ x_t,      // [B][NIN] bf16
    const u16* __restrict__ spk_in,   // [B][NH]  bf16 (binary)
    u16*       __restrict__ spk_out,  // [B][NH]  bf16
    const u16* __restrict__ Ws,       // [NS][KIN] bf16
    const u16* __restrict__ Wa,       // [NA][KIN] bf16
    const float* __restrict__ bias_s,
    const float* __restrict__ bias_a,
    float* __restrict__ vs,           // [B][NS]
    float* __restrict__ va,           // [B][NA]
    float* __restrict__ bstate)       // [B][NA]
{
  __shared__ u16 Bs[2][64 * BSTRIDE];   // 2 x 5120 B staged weight tiles

  const int tid  = threadIdx.x;
  const int lane = tid & 31;
  const int wave = tid >> 5;
  const int half = lane >> 4;
  const int l15  = lane & 15;

  const int mb  = blockIdx.x * 64 + (wave >> 1) * 16;  // batch base of this wave
  const int ngB = blockIdx.y * 64;                     // block neuron base [0,2048)
  const int ng  = ngB + (wave & 1) * 32;               // wave neuron base
  const bool adapt = (ngB >= NS);                      // 64-wide block tile on one side
  const u16* Wb  = adapt ? Wa : Ws;
  const int  nwB = adapt ? (ngB - NS) : ngB;           // block row base within W

  // A per-lane row pointers (contiguous k = 16*half + s, same as B; 16B aligned)
  const u16* pa_x = x_t    + (mb + l15) * NIN;
  const u16* pa_s = spk_in + (mb + l15) * NH;

  // staging assignment: thread -> (row 0..63, 16B chunk 0..3) of the 64x32 tile
  const int srow = tid >> 2;
  const int sc   = (tid & 3) * 8;                      // u16 offset in chunk
  const u16* gB  = Wb + (nwB + srow) * KIN + sc;       // + k per step
  const int  lofs = srow * BSTRIDE + sc;               // LDS u16 offset in a buffer

  // B fragment read base for this wave: local rows (wave&1)*32 + l15 (+16)
  const int rd0 = ((wave & 1) * 32 + l15) * BSTRIDE + half * 16;
  const int rd1 = rd0 + 16 * BSTRIDE;

#if __has_builtin(__builtin_amdgcn_global_load_async_to_lds_b128)
  #define ASYNC_STAGE 1
  __builtin_amdgcn_global_load_async_to_lds_b128(
      (AS1 v4i*)gB, (AS3 v4i*)&Bs[0][lofs], 0, 0);
#else
  #define ASYNC_STAGE 0
  u16x8 pend = *(const u16x8*)gB;                      // prefetch chunk k=0
#endif

  f32x8 acc0 = {0.f,0.f,0.f,0.f,0.f,0.f,0.f,0.f};
  f32x8 acc1 = acc0;

  for (int k = 0; k < KIN; k += 32) {
    const int buf = (k >> 5) & 1;
#if ASYNC_STAGE
    wait_async0();                      // this wave's staged chunk is in LDS
    __syncthreads();                    // all waves staged + done reading buf^1
    if (k + 32 < KIN)
      __builtin_amdgcn_global_load_async_to_lds_b128(
          (AS1 v4i*)(gB + k + 32), (AS3 v4i*)&Bs[buf ^ 1][lofs], 0, 0);
#else
    __syncthreads();                    // done reading Bs[buf] from iter k-64
    *(u16x8*)&Bs[buf][lofs] = pend;
    __syncthreads();                    // staged chunk visible
    if (k + 32 < KIN) pend = *(const u16x8*)(gB + k + 32);
#endif

    const u16* paRow = (k < NIN) ? (pa_x + k) : (pa_s + (k - NIN));
    Frag a, b0, b1;
    a.h[0]  = *(const u16x8*)(paRow + half * 16);
    a.h[1]  = *(const u16x8*)(paRow + half * 16 + 8);
    b0.h[0] = *(const u16x8*)&Bs[buf][rd0];
    b0.h[1] = *(const u16x8*)&Bs[buf][rd0 + 8];
    b1.h[0] = *(const u16x8*)&Bs[buf][rd1];
    b1.h[1] = *(const u16x8*)&Bs[buf][rd1 + 8];
    acc0 = wmma_bf16(a, b0, acc0);
    acc1 = wmma_bf16(a, b1, acc1);
  }

  // fused LIF epilogue. C/D layout: vgpr r, lanes 0-15 -> M=r, lanes 16-31 -> M=8+r; N=lane%16
  const int row0 = mb + half * 8;
#pragma unroll
  for (int j = 0; j < 2; ++j) {
    f32x8 acc = j ? acc1 : acc0;
    const int n  = ng + l15 + j * 16;          // global neuron [0,2048)
    const int nl = adapt ? (n - NS) : n;       // local neuron within population
    const float bia = adapt ? bias_a[nl] : bias_s[nl];
#pragma unroll
    for (int r = 0; r < 8; ++r) {
      const int b    = row0 + r;
      const int sidx = b * NS + nl;            // NS == NA == 1024
      const float h  = acc[r] + bia;
      float s;
      if (!adapt) {
        float v = ALPHA * vs[sidx] + h;
        s = (v > THRESH) ? 1.f : 0.f;          // spike((v-TH)/TH), TH=1
        vs[sidx] = v - THRESH * s;
      } else {
        float bav = bstate[sidx];
        float th  = THRESH + BETA * bav;
        float v   = ALPHA * va[sidx] + h;
        s = (v > th) ? 1.f : 0.f;
        va[sidx]     = v - th * s;
        bstate[sidx] = RHO * bav + (1.f - RHO) * s;
      }
      spk_out[b * NH + n] = (s > 0.f) ? (u16)0x3F80 : (u16)0;  // bf16 1.0 / 0.0 exact
    }
  }
}

// ---------------- readout GEMM + leaky integrator ----------------
// vo = (first ? 0 : ALPHA_OUT*vo) + spk @ Wo^T + bo ; write out on last inner step.
// grid (2, 4); block 256 = 8 waves; wave tile 16 x 32. Direct-from-global (~9% of FLOPs).
__global__ __launch_bounds__(256) void rsnn_readout(
    const u16* __restrict__ spk,   // [B][NH] bf16
    const u16* __restrict__ Wo,    // [NOUT][NH] bf16
    const float* __restrict__ bo,
    float* __restrict__ vo,        // [B][NOUT]
    float* __restrict__ out_t,     // [B][NOUT]
    int first, int last)
{
  const int lane = threadIdx.x & 31;
  const int wave = threadIdx.x >> 5;
  const int half = lane >> 4;
  const int l15  = lane & 15;

  const int mb = blockIdx.x * 64 + (wave >> 1) * 16;
  const int nb = blockIdx.y * 64 + (wave & 1) * 32;

  const u16* pa  = spk + (mb + l15) * NH;
  const u16* pb0 = Wo + (nb + l15) * NH;
  const u16* pb1 = pb0 + 16 * NH;

  f32x8 acc0 = {0.f,0.f,0.f,0.f,0.f,0.f,0.f,0.f};
  f32x8 acc1 = acc0;
  Frag a, b0, b1;

  for (int k = 0; k < NH; k += 32) {
    a.h[0]  = *(const u16x8*)(pa + k + half * 16);
    a.h[1]  = *(const u16x8*)(pa + k + half * 16 + 8);
    b0.h[0] = *(const u16x8*)(pb0 + k + half * 16);
    b0.h[1] = *(const u16x8*)(pb0 + k + half * 16 + 8);
    b1.h[0] = *(const u16x8*)(pb1 + k + half * 16);
    b1.h[1] = *(const u16x8*)(pb1 + k + half * 16 + 8);
    acc0 = wmma_bf16(a, b0, acc0);
    acc1 = wmma_bf16(a, b1, acc1);
  }

  const int row0 = mb + half * 8;
#pragma unroll
  for (int j = 0; j < 2; ++j) {
    f32x8 acc = j ? acc1 : acc0;
    const int n = nb + l15 + j * 16;
    const float bia = bo[n];
#pragma unroll
    for (int r = 0; r < 8; ++r) {
      const int b   = row0 + r;
      const int idx = b * NOUT + n;
      float prev = first ? 0.f : vo[idx];
      float v = ALPHA_OUT * prev + acc[r] + bia;
      vo[idx] = v;
      if (last) out_t[idx] = v;
    }
  }
}

// ---------------- host driver ----------------
extern "C" void kernel_launch(void* const* d_in, const int* in_sizes, int n_in,
                              void* d_out, int out_size, void* d_ws, size_t ws_size,
                              hipStream_t stream) {
  (void)in_sizes; (void)n_in; (void)out_size; (void)ws_size;
  const float* inp = (const float*)d_in[0];  // [T][B][NIN]
  const float* W_s = (const float*)d_in[1];  // [NS][KIN]
  const float* b_s = (const float*)d_in[2];
  const float* W_a = (const float*)d_in[3];  // [NA][KIN]
  const float* b_a = (const float*)d_in[4];
  const float* W_o = (const float*)d_in[5];  // [NOUT][NH]
  const float* b_o = (const float*)d_in[6];
  float* out = (float*)d_out;                // [T][B][NOUT]

  char* ws = (char*)d_ws;
  size_t off = 0;
  auto take = [&](size_t bytes) -> char* {
    char* p = ws + off;
    off += (bytes + 255) & ~(size_t)255;
    return p;
  };

  u16* xin   = (u16*)take((size_t)T * B * NIN * 2);   // bf16 input, converted once
  u16* Wsb   = (u16*)take((size_t)NS * KIN * 2);
  u16* Wab   = (u16*)take((size_t)NA * KIN * 2);
  u16* Wob   = (u16*)take((size_t)NOUT * NH * 2);
  u16* spk0  = (u16*)take((size_t)B * NH * 2);        // double-buffered spikes
  u16* spk1  = (u16*)take((size_t)B * NH * 2);
  float* vs  = (float*)take((size_t)B * NS * 4);
  float* va  = (float*)take((size_t)B * NA * 4);
  float* bst = (float*)take((size_t)B * NA * 4);
  float* vo  = (float*)take((size_t)B * NOUT * 4);

  // one-time (per call) precision conversion; weights then live in L2 for all 200 steps
  cvt_f32_bf16<<<1024, 256, 0, stream>>>(inp, xin, T * B * NIN);
  cvt_f32_bf16<<<1024, 256, 0, stream>>>(W_s, Wsb, NS * KIN);
  cvt_f32_bf16<<<1024, 256, 0, stream>>>(W_a, Wab, NA * KIN);
  cvt_f32_bf16<<<256, 256, 0, stream>>>(W_o, Wob, NOUT * NH);

  // zero-init recurrent state (graph-capture-safe memsets)
  (void)hipMemsetAsync(spk0, 0, (size_t)B * NH * 2, stream);
  (void)hipMemsetAsync(vs,   0, (size_t)B * NS * 4, stream);
  (void)hipMemsetAsync(va,   0, (size_t)B * NA * 4, stream);
  (void)hipMemsetAsync(bst,  0, (size_t)B * NA * 4, stream);

  u16* sbuf[2] = {spk0, spk1};
  int cur = 0;
  for (int t = 0; t < T; ++t) {
    const u16* x_t = xin + (size_t)t * B * NIN;
    for (int s = 0; s < SIM; ++s) {
      rsnn_hidden<<<dim3(2, 32), 256, 0, stream>>>(
          x_t, sbuf[cur], sbuf[cur ^ 1], Wsb, Wab, b_s, b_a, vs, va, bst);
      rsnn_readout<<<dim3(2, 4), 256, 0, stream>>>(
          sbuf[cur ^ 1], Wob, b_o, vo, out + (size_t)t * B * NOUT,
          (s == 0) ? 1 : 0, (s == SIM - 1) ? 1 : 0);
      cur ^= 1;
    }
  }
}
